// NeuralFieldCosmo_59820304499244
// MI455X (gfx1250) — matrix operand order repaired and checked by reading, not compiled, not run on hardware
//
#include <hip/hip_runtime.h>
#include <hip/hip_bf16.h>

typedef __attribute__((ext_vector_type(16))) _Float16 v16h;
typedef __attribute__((ext_vector_type(8)))  float    v8f;

#define N_NODES 10000
#define N_EDGES 160000
#define FC      32
#define IN_C    32
#define OUT_C   32
#define LN_EPS  1e-5f
#define NWAVES  8
#define THREADS 256
// workspace layout (floats): sums[320000] | counts[10000] | Bpack (uint, 16B aligned)
#define WS_SUMS   0
#define WS_COUNTS 320000
#define WS_BPACK  330000   // 330000*4 = 1,320,000 bytes, 16B aligned

union ABFrag { uint4 u4[2]; v16h v; unsigned u[8]; };

__device__ __forceinline__ unsigned pack2h(float a, float b) {
  union { _Float16 h[2]; unsigned u; } p;
  p.h[0] = (_Float16)a; p.h[1] = (_Float16)b;
  return p.u;
}

// Hardware tanh (CDNA5 V_TANH_F32 transcendental).
__device__ __forceinline__ float htanh(float x) {
#if __has_builtin(__builtin_amdgcn_tanhf)
  return __builtin_amdgcn_tanhf(x);
#else
  float r;
  asm volatile("v_tanh_f32 %0, %1" : "=v"(r) : "v"(x));
  return r;
#endif
}

// 16-lane xor-reduction stage via DPP16 ROW_XMASK (0x160|mask): pure VALU.
template <int MASK>
__device__ __forceinline__ float dpp_xadd(float x) {
  int other = __builtin_amdgcn_update_dpp(0, __float_as_int(x),
                                          0x160 | MASK, 0xf, 0xf, true);
  return x + __int_as_float(other);
}

__device__ __forceinline__ float rowsum16(float s) {
  s = dpp_xadd<1>(s);
  s = dpp_xadd<2>(s);
  s = dpp_xadd<4>(s);
  s = dpp_xadd<8>(s);
  return s;
}

// DPP16 ROW_SHL:8 (0x100|8): moves lanes {0-7,16-23} into {8-15,24-31}.
__device__ __forceinline__ float dpp_shl8(float x) {
  int r = __builtin_amdgcn_update_dpp(0, __float_as_int(x), 0x108, 0xf, 0xf, true);
  return __int_as_float(r);
}

// --- pack W2 (32x1024 f32, row-major K x N) into f16 B-fragment order ---
__global__ __launch_bounds__(256) void pack_w2_kernel(const float* __restrict__ W2,
                                                      unsigned* __restrict__ Bpack) {
  int tid = blockIdx.x * 256 + threadIdx.x;     // 0 .. 16383
  int t = tid >> 8;
  int q = tid & 255;
  int l = q >> 3;
  int v = q & 7;
  int n = l & 15;
  int kbase = (l & 16) ? 16 : 0;
  int K0 = kbase + 2 * v;
  int col = t * 16 + n;
  float f0 = W2[K0 * 1024 + col];
  float f1 = W2[(K0 + 1) * 1024 + col];
  Bpack[tid] = pack2h(f0, f1);
}

__global__ __launch_bounds__(256) void zero_kernel(float* __restrict__ p, int n) {
  int i = blockIdx.x * 256 + threadIdx.x;
  if (i < n) p[i] = 0.0f;
}

__global__ __launch_bounds__(256) void finalize_kernel(const float* __restrict__ sums,
                                                       const float* __restrict__ counts,
                                                       float* __restrict__ out) {
  int i = blockIdx.x * 256 + threadIdx.x;
  if (i < N_NODES * OUT_C) {
    float c = counts[i >> 5];
    out[i] = sums[i] / fmaxf(c, 1.0f);
  }
}

// --- fused neural-field + gather-matvec + scatter-sum kernel ---
__global__ __launch_bounds__(THREADS) void nf_main_kernel(
    const float* __restrict__ features, const float* __restrict__ hood,
    const int* __restrict__ source, const int* __restrict__ target,
    const float* __restrict__ W0, const float* __restrict__ b0,
    const float* __restrict__ g0, const float* __restrict__ be0,
    const float* __restrict__ W1, const float* __restrict__ b1,
    const float* __restrict__ g1, const float* __restrict__ be1,
    const float* __restrict__ b2, const unsigned* __restrict__ Bpack,
    float* __restrict__ sums, float* __restrict__ counts) {
  __shared__ uint4 Bsh[4096];          // 64 KB: f16 W2 in B-fragment order
  __shared__ float b2sh[1024];         // 4 KB
  __shared__ float W0sh[96], b0sh[32], g0sh[32], be0sh[32];
  __shared__ float W1sh[1024], b1sh[32], g1sh[32], be1sh[32];
  __shared__ unsigned hsh[NWAVES][512];   // 16 KB: per-wave 32 edges x 32ch packed f16
  __shared__ float    fsh[NWAVES][1024];  // 32 KB: per-wave gathered features f32
  __shared__ int      tsh[NWAVES][32];    // per-wave targets

  const int tid = threadIdx.x;
  const int w   = tid >> 5;   // wave in block
  const int l   = tid & 31;   // lane in wave

  // cooperative parameter staging
  const uint4* Bp4 = (const uint4*)Bpack;
  #pragma unroll
  for (int i = 0; i < 16; ++i) Bsh[tid + 256 * i] = Bp4[tid + 256 * i];
  #pragma unroll
  for (int i = 0; i < 4; ++i) {
    b2sh[tid + 256 * i] = b2[tid + 256 * i];
    W1sh[tid + 256 * i] = W1[tid + 256 * i];
  }
  if (tid < 96) W0sh[tid] = W0[tid];
  if (tid < 32) {
    b0sh[tid] = b0[tid]; g0sh[tid] = g0[tid]; be0sh[tid] = be0[tid];
    b1sh[tid] = b1[tid]; g1sh[tid] = g1[tid]; be1sh[tid] = be1[tid];
  }
  __syncthreads();

  // ---- per-lane edge MLP: 3 -> 32 -> LN -> relu -> 32 -> LN -> relu ----
  const int e = blockIdx.x * THREADS + tid;    // 625 blocks * 256 = 160000 exact
  const float x0 = hood[3 * e + 0];
  const float x1 = hood[3 * e + 1];
  const float x2 = hood[3 * e + 2];

  float h[FC];
  #pragma unroll
  for (int c = 0; c < FC; ++c)
    h[c] = fmaf(x0, W0sh[c], fmaf(x1, W0sh[32 + c], fmaf(x2, W0sh[64 + c], b0sh[c])));
  {
    float mu = 0.f;
    #pragma unroll
    for (int c = 0; c < FC; ++c) mu += h[c];
    mu *= (1.0f / FC);
    float var = 0.f;
    #pragma unroll
    for (int c = 0; c < FC; ++c) { float d = h[c] - mu; var = fmaf(d, d, var); }
    float inv = rsqrtf(var * (1.0f / FC) + LN_EPS);
    #pragma unroll
    for (int c = 0; c < FC; ++c)
      h[c] = fmaxf(0.f, fmaf((h[c] - mu) * inv, g0sh[c], be0sh[c]));
  }
  float h2[FC];
  #pragma unroll
  for (int c = 0; c < FC; ++c) h2[c] = b1sh[c];
  #pragma unroll
  for (int k = 0; k < FC; ++k) {
    float hk = h[k];
    #pragma unroll
    for (int c = 0; c < FC; ++c) h2[c] = fmaf(hk, W1sh[k * 32 + c], h2[c]);
  }
  {
    float mu = 0.f;
    #pragma unroll
    for (int c = 0; c < FC; ++c) mu += h2[c];
    mu *= (1.0f / FC);
    float var = 0.f;
    #pragma unroll
    for (int c = 0; c < FC; ++c) { float d = h2[c] - mu; var = fmaf(d, d, var); }
    float inv = rsqrtf(var * (1.0f / FC) + LN_EPS);
    #pragma unroll
    for (int c = 0; c < FC; ++c)
      h2[c] = fmaxf(0.f, fmaf((h2[c] - mu) * inv, g1sh[c], be1sh[c]));
  }

  // stage h (packed f16), gathered features, target
  #pragma unroll
  for (int d = 0; d < 16; ++d) hsh[w][l * 16 + d] = pack2h(h2[2 * d], h2[2 * d + 1]);
  const int src = source[e];
  const int tgt = target[e];
  tsh[w][l] = tgt;
  const float4* fsrc = (const float4*)(features + src * IN_C);
  float4* fdst = (float4*)&fsh[w][l * 32];
  #pragma unroll
  for (int d = 0; d < 8; ++d) fdst[d] = fsrc[d];
  atomicAdd(&counts[tgt], 1.0f);
  __syncthreads();

  // ---- WMMA phase: [32 edges x 32] @ [32 x 1024] in 2 M-tiles x 64 N-tiles ----
  const int n    = l & 15;
  const int half = l >> 4;

  // A fragments (16-bit 16x32 layout)
  ABFrag a[2];
  #pragma unroll
  for (int mt = 0; mt < 2; ++mt) {
    int r = mt * 16 + n;
    int base = r * 16 + half * 8;
    a[mt].u4[0] = *(const uint4*)&hsh[w][base];
    a[mt].u4[1] = *(const uint4*)&hsh[w][base + 4];
  }

  // o-invariant epilogue operands
  float fr0[2][8], fr1[2][8];
  #pragma unroll
  for (int mt = 0; mt < 2; ++mt) {
    #pragma unroll
    for (int v = 0; v < 8; ++v) {
      int r = mt * 16 + v + half * 8;            // edge row within wave's 32
      fr0[mt][v] = fsh[w][r * 32 + n];
      fr1[mt][v] = fsh[w][r * 32 + 16 + n];
    }
  }
  // Per-lane scatter pointer for the merged row-sum register:
  //   lane -> row mapping after pack+shl8 merge: row = 8*half + (n&7) + (n&8 ? 16 : 0)
  const int rlane = 8 * half + (n & 7) + ((n & 8) ? 16 : 0);
  float* const scat = sums + tsh[w][rlane] * OUT_C;

  #pragma unroll 2
  for (int o = 0; o < OUT_C; ++o) {
    const int t0 = 2 * o, t1 = 2 * o + 1;      // two 16-col tiles covering i=0..31 for output o
    ABFrag bf0, bf1;
    bf0.u4[0] = Bsh[t0 * 64 + l * 2];
    bf0.u4[1] = Bsh[t0 * 64 + l * 2 + 1];
    bf1.u4[0] = Bsh[t1 * 64 + l * 2];
    bf1.u4[1] = Bsh[t1 * 64 + l * 2 + 1];
    const float bias0 = b2sh[t0 * 16 + n];
    const float bias1 = b2sh[t1 * 16 + n];

    float pk[2];
    #pragma unroll
    for (int mt = 0; mt < 2; ++mt) {
      v8f c0 = {}, c1 = {};
      c0 = __builtin_amdgcn_wmma_f32_16x16x32_f16(false, a[mt].v, false, bf0.v,
                                                  (short)0, c0, false, false);
      c1 = __builtin_amdgcn_wmma_f32_16x16x32_f16(false, a[mt].v, false, bf1.v,
                                                  (short)0, c1, false, false);
      float p = 0.f;
      #pragma unroll
      for (int v = 0; v < 8; ++v) {
        float s = htanh(c0[v] + bias0) * fr0[mt][v] + htanh(c1[v] + bias1) * fr1[mt][v];
        s = rowsum16(s);             // row-sum replicated across each 16-lane half
        p = (n == v) ? s : p;        // pack row-sums: lane v <- row v, lane 16+v <- row v+8
      }
      pk[mt] = p;
    }
    // merge two packed registers into 32 lanes (mt=1 shifted into lanes {8-15,24-31})
    float merged = (n < 8) ? pk[0] : dpp_shl8(pk[1]);
    // single full-EXEC vector atomic scatters all 32 row-sums for this o
    atomicAdd(scat + o, merged);
  }
}

extern "C" void kernel_launch(void* const* d_in, const int* in_sizes, int n_in,
                              void* d_out, int out_size, void* d_ws, size_t ws_size,
                              hipStream_t stream) {
  const float* features = (const float*)d_in[0];
  const float* hood     = (const float*)d_in[1];
  const int*   source   = (const int*)d_in[2];
  const int*   target   = (const int*)d_in[3];
  const float* W0  = (const float*)d_in[4];
  const float* b0  = (const float*)d_in[5];
  const float* g0  = (const float*)d_in[6];
  const float* be0 = (const float*)d_in[7];
  const float* W1  = (const float*)d_in[8];
  const float* b1  = (const float*)d_in[9];
  const float* g1  = (const float*)d_in[10];
  const float* be1 = (const float*)d_in[11];
  const float* W2  = (const float*)d_in[12];
  const float* b2  = (const float*)d_in[13];

  float*    ws     = (float*)d_ws;
  float*    sums   = ws + WS_SUMS;
  float*    counts = ws + WS_COUNTS;
  unsigned* Bpack  = (unsigned*)(ws + WS_BPACK);

  pack_w2_kernel<<<64, 256, 0, stream>>>(W2, Bpack);
  zero_kernel<<<(330000 + 255) / 256, 256, 0, stream>>>(ws, 330000); // sums+counts
  nf_main_kernel<<<N_EDGES / THREADS, THREADS, 0, stream>>>(
      features, hood, source, target, W0, b0, g0, be0, W1, b1, g1, be1,
      b2, Bpack, sums, counts);
  finalize_kernel<<<(N_NODES * OUT_C + 255) / 256, 256, 0, stream>>>(
      sums, counts, (float*)d_out);
}
